// RN_29137058136796
// MI455X (gfx1250) — matrix operand order repaired and checked by reading, not compiled, not run on hardware
//
#include <hip/hip_runtime.h>
#include <stdint.h>

// ---------------------------------------------------------------------------
// Relation Network forward for MI455X (gfx1250, wave32, WMMA).
// Heavy path: 320000x{64,256} @ {64,256}x256 GEMMs in bf16 via
// v_wmma_f32_16x16x32_bf16, LDS double-buffered tiles, f32 accumulate.
// Grid puts N-blocks fastest so the A tile is reused out of L2 (192 MB).
// ---------------------------------------------------------------------------

typedef __attribute__((ext_vector_type(16))) __bf16          v16bf;
typedef __attribute__((ext_vector_type(16))) unsigned short  v16u;
typedef __attribute__((ext_vector_type(8)))  float           v8f;

#define MBB   512
#define NLOC  25
#define QSTD  11
#define EPSV  1e-5f

__device__ __forceinline__ unsigned short f2bf(float f) {
  unsigned int u = __float_as_uint(f);
  u += 0x7fffu + ((u >> 16) & 1u);          // round-to-nearest-even
  return (unsigned short)(u >> 16);
}
__device__ __forceinline__ float bf2f(unsigned short h) {
  return __uint_as_float(((unsigned int)h) << 16);
}

// ---------------------------------------------------------------------------
// Conv2d(k=3, s=2, p=1) + bias + ReLU, fp32. Cout fixed at 24.
// ---------------------------------------------------------------------------
__global__ void conv_relu_k(const float* __restrict__ in, const float* __restrict__ w,
                            const float* __restrict__ bias, float* __restrict__ out,
                            int Cin, int Hin, int Win, int Hout, int Wout) {
  int idx = blockIdx.x * blockDim.x + threadIdx.x;
  int total = MBB * 24 * Hout * Wout;
  if (idx >= total) return;
  int wo = idx % Wout; int t = idx / Wout;
  int ho = t % Hout;   t /= Hout;
  int co = t % 24;     int b = t / 24;
  float acc = bias[co];
  for (int ci = 0; ci < Cin; ++ci) {
    const float* ip = in + ((b * Cin + ci) * Hin) * Win;
    const float* wp = w + ((co * Cin + ci) * 3) * 3;
#pragma unroll
    for (int kh = 0; kh < 3; ++kh) {
      int hi = ho * 2 - 1 + kh;
      if (hi < 0 || hi >= Hin) continue;
#pragma unroll
      for (int kw = 0; kw < 3; ++kw) {
        int wi = wo * 2 - 1 + kw;
        if (wi < 0 || wi >= Win) continue;
        acc += ip[hi * Win + wi] * wp[kh * 3 + kw];
      }
    }
  }
  out[idx] = fmaxf(acc, 0.f);
}

// Per-channel batch stats (mean, rsqrt(var+eps)). One block per channel.
__global__ void bn_stats_k(const float* __restrict__ y, float* __restrict__ stats, int HW) {
  __shared__ float ss[256], ss2[256];
  int c = blockIdx.x, tid = threadIdx.x;
  float s = 0.f, s2 = 0.f;
  int total = MBB * HW;
  for (int i = tid; i < total; i += 256) {
    int b = i / HW, hw = i - b * HW;
    float v = y[(b * 24 + c) * HW + hw];
    s += v; s2 += v * v;
  }
  ss[tid] = s; ss2[tid] = s2;
  __syncthreads();
  for (int st = 128; st > 0; st >>= 1) {
    if (tid < st) { ss[tid] += ss[tid + st]; ss2[tid] += ss2[tid + st]; }
    __syncthreads();
  }
  if (tid == 0) {
    float inv  = 1.f / (float)total;
    float mean = ss[0] * inv;
    float var  = ss2[0] * inv - mean * mean;
    stats[c * 2]     = mean;
    stats[c * 2 + 1] = rsqrtf(var + EPSV);
  }
}

__global__ void bn_apply_k(float* __restrict__ y, const float* __restrict__ stats,
                           const float* __restrict__ g, const float* __restrict__ beta, int HW) {
  int idx = blockIdx.x * blockDim.x + threadIdx.x;
  int total = MBB * 24 * HW;
  if (idx >= total) return;
  int c = (idx / HW) % 24;
  y[idx] = (y[idx] - stats[c * 2]) * stats[c * 2 + 1] * g[c] + beta[c];
}

// fp32 [rows x kin] weights -> bf16 [rows x kout], zero-padded K.
__global__ void wconv_k(const float* __restrict__ src, unsigned short* __restrict__ dst,
                        int rows, int kin, int kout) {
  int idx = blockIdx.x * blockDim.x + threadIdx.x;
  if (idx >= rows * kout) return;
  int r = idx / kout, k = idx - r * kout;
  dst[idx] = (k < kin) ? f2bf(src[r * kin + k]) : (unsigned short)0;
}

// ---------------------------------------------------------------------------
// Build pair features: P[((b*25+a)*25+c), k] bf16, k in [0,64)
// ---------------------------------------------------------------------------
__global__ void pair_k(const float* __restrict__ x4, const float* __restrict__ qst,
                       unsigned short* __restrict__ P) {
  int idx = blockIdx.x * blockDim.x + threadIdx.x;   // row*64 + k
  if (idx >= MBB * 625 * 64) return;
  int k   = idx & 63;
  int row = idx >> 6;
  int c = row % 25;
  int t = row / 25;
  int a = t % 25;
  int b = t / 25;
  float v;
  if (k < 26) {
    if (k < 24)       v = x4[(b * 24 + k) * 25 + c];
    else if (k == 24) v = ((float)c / 5.0f - 2.0f) * 0.5f;
    else              v = ((float)(c % 5) - 2.0f) * 0.5f;
  } else if (k < 52) {
    int kk = k - 26;
    if (kk < 24)       v = x4[(b * 24 + kk) * 25 + a];
    else if (kk == 24) v = ((float)a / 5.0f - 2.0f) * 0.5f;
    else               v = ((float)(a % 5) - 2.0f) * 0.5f;
  } else if (k < 63) {
    v = qst[b * QSTD + (k - 52)];
  } else {
    v = 0.f;
  }
  P[idx] = f2bf(v);
}

// ---------------------------------------------------------------------------
// C[M,N] = act(A[MxK] @ W[NxK]^T + bias), bf16 in/out, f32 accumulate.
// Block tile 128x128, K-step 32, 256 threads = 8 waves (4 along M x 2 along N),
// each wave: 32x64 output = 2x4 v_wmma_f32_16x16x32_bf16 tiles.
// LDS double-buffered: next tile's global loads overlap current WMMAs.
// Requires M%128==0, N%128==0, K%32==0 (true for all call sites).
// blockIdx.x = N block (fast, 2 blocks) so A tiles hit L2 across N blocks.
// ---------------------------------------------------------------------------
#define BM 128
#define BN 128
#define BK 32
#define LDSROW 40   // halfword stride: 80 B, keeps 16B alignment, bank-shifts rows

__global__ __launch_bounds__(256) void gemm_bias_relu(
    const unsigned short* __restrict__ A,
    const unsigned short* __restrict__ W,
    const float* __restrict__ bias,
    unsigned short* __restrict__ C,
    int M, int N, int K, int relu) {
  __shared__ unsigned short As[2][BM * LDSROW];
  __shared__ unsigned short Ws[2][BN * LDSROW];

  int tid  = threadIdx.x;
  int lane = tid & 31;
  int wid  = tid >> 5;
  int wm   = wid & 3;               // wave's M patch: wm*32
  int wn   = wid >> 2;              // wave's N patch: wn*64
  int n0   = blockIdx.x * BN;       // N fastest -> A tile L2 reuse
  int m0   = blockIdx.y * BM;

  // staging coordinates: 128x32 tile = 512 x 16B vectors, 2 per thread
  int srow0 = tid >> 2,          scv0 = tid & 3;
  int srow1 = (tid + 256) >> 2,  scv1 = (tid + 256) & 3;

  uint4 ra0, ra1, rb0, rb1;
  const unsigned short* Abase = A + m0 * K;
  const unsigned short* Wbase = W + n0 * K;

#define G2R(k0)                                                                 \
  do {                                                                          \
    ra0 = *reinterpret_cast<const uint4*>(Abase + srow0 * K + (k0) + scv0 * 8); \
    ra1 = *reinterpret_cast<const uint4*>(Abase + srow1 * K + (k0) + scv1 * 8); \
    rb0 = *reinterpret_cast<const uint4*>(Wbase + srow0 * K + (k0) + scv0 * 8); \
    rb1 = *reinterpret_cast<const uint4*>(Wbase + srow1 * K + (k0) + scv1 * 8); \
  } while (0)

#define R2S(buf)                                                                  \
  do {                                                                            \
    *reinterpret_cast<uint4*>(&As[buf][srow0 * LDSROW + scv0 * 8]) = ra0;         \
    *reinterpret_cast<uint4*>(&As[buf][srow1 * LDSROW + scv1 * 8]) = ra1;         \
    *reinterpret_cast<uint4*>(&Ws[buf][srow0 * LDSROW + scv0 * 8]) = rb0;         \
    *reinterpret_cast<uint4*>(&Ws[buf][srow1 * LDSROW + scv1 * 8]) = rb1;         \
  } while (0)

  v8f zero = {};
  v8f acc[2][4];
#pragma unroll
  for (int i = 0; i < 2; ++i)
#pragma unroll
    for (int j = 0; j < 4; ++j) acc[i][j] = zero;

  int mfl   = lane & 15;
  int khalf = lane >> 4;

  G2R(0);
  R2S(0);
  int cur = 0;

  for (int k0 = 0; k0 < K; k0 += BK) {
    __syncthreads();                       // LDS[cur] ready for all waves
    bool more = (k0 + BK) < K;
    if (more) G2R(k0 + BK);                // global loads overlap WMMA below
    if (k0 + 2 * BK < K) {                 // L2 prefetch of the tile after next
      __builtin_prefetch(Abase + srow0 * K + k0 + 2 * BK, 0, 2);
      __builtin_prefetch(Wbase + srow0 * K + k0 + 2 * BK, 0, 2);
    }

    // --- gather WMMA fragments from LDS (ISA 7.12.2 lane layouts) ---
    v16u aU[2], bU[4];
#pragma unroll
    for (int i = 0; i < 2; ++i) {
      int m = wm * 32 + i * 16 + mfl;
#pragma unroll
      for (int e = 0; e < 16; ++e) {
        int k = e + 8 * khalf + ((e >= 8) ? 8 : 0);
        aU[i][e] = As[cur][m * LDSROW + k];
      }
    }
#pragma unroll
    for (int j = 0; j < 4; ++j) {
      int n = wn * 64 + j * 16 + mfl;
#pragma unroll
      for (int e = 0; e < 16; ++e)
        bU[j][e] = Ws[cur][n * LDSROW + e + 8 * khalf + ((e >= 8) ? 8 : 0)];
    }

#pragma unroll
    for (int i = 0; i < 2; ++i)
#pragma unroll
      for (int j = 0; j < 4; ++j)
        acc[i][j] = __builtin_amdgcn_wmma_f32_16x16x32_bf16(
            false, __builtin_bit_cast(v16bf, aU[i]),
            false, __builtin_bit_cast(v16bf, bU[j]),
            (short)0, acc[i][j], false, false);

    if (more) R2S(cur ^ 1);                // deposit next tile into other buffer
    cur ^= 1;
  }

  // --- epilogue: bias + relu + bf16 store ---
#pragma unroll
  for (int j = 0; j < 4; ++j) {
    int n = n0 + wn * 64 + j * 16 + mfl;
    float bv = bias[n];
#pragma unroll
    for (int i = 0; i < 2; ++i) {
#pragma unroll
      for (int r = 0; r < 8; ++r) {
        int m = m0 + wm * 32 + i * 16 + r + 8 * khalf;
        float v = acc[i][j][r] + bv;
        if (relu) v = fmaxf(v, 0.f);
        C[m * N + n] = f2bf(v);
      }
    }
  }
#undef G2R
#undef R2S
}

// Sum over 625 pairs: xg[b,n] = sum_p h[(b*625+p), n]
__global__ void pool_k(const unsigned short* __restrict__ H, unsigned short* __restrict__ xg) {
  int idx = blockIdx.x * blockDim.x + threadIdx.x;
  if (idx >= MBB * 256) return;
  int b = idx >> 8, n = idx & 255;
  const unsigned short* p = H + ((long)b * 625) * 256 + n;
  float s = 0.f;
  for (int i = 0; i < 625; ++i) s += bf2f(p[(long)i * 256]);
  xg[idx] = f2bf(s);
}

// Final FC 256->10 + log_softmax. One wave32 per batch row.
__global__ void final_k(const unsigned short* __restrict__ y2, const float* __restrict__ fw3,
                        const float* __restrict__ fb3, float* __restrict__ out) {
  int b = blockIdx.x;
  int lane = threadIdx.x;
  float acc[10];
#pragma unroll
  for (int o = 0; o < 10; ++o) acc[o] = 0.f;
  for (int k = lane; k < 256; k += 32) {
    float x = bf2f(y2[b * 256 + k]);
#pragma unroll
    for (int o = 0; o < 10; ++o) acc[o] += x * fw3[o * 256 + k];
  }
#pragma unroll
  for (int o = 0; o < 10; ++o)
    for (int off = 16; off > 0; off >>= 1)
      acc[o] += __shfl_xor(acc[o], off, 32);
  if (lane == 0) {
    float y[10], mx = -1e30f;
#pragma unroll
    for (int o = 0; o < 10; ++o) { y[o] = acc[o] + fb3[o]; mx = fmaxf(mx, y[o]); }
    float s = 0.f;
#pragma unroll
    for (int o = 0; o < 10; ++o) s += expf(y[o] - mx);
    float lse = mx + logf(s);
#pragma unroll
    for (int o = 0; o < 10; ++o) out[b * 10 + o] = y[o] - lse;
  }
}

// ---------------------------------------------------------------------------
extern "C" void kernel_launch(void* const* d_in, const int* in_sizes, int n_in,
                              void* d_out, int out_size, void* d_ws, size_t ws_size,
                              hipStream_t stream) {
  (void)in_sizes; (void)n_in; (void)out_size; (void)ws_size;

  const float* img = (const float*)d_in[0];
  const float* qst = (const float*)d_in[1];
  const float* cw[4]  = {(const float*)d_in[2], (const float*)d_in[6],
                         (const float*)d_in[10], (const float*)d_in[14]};
  const float* cb[4]  = {(const float*)d_in[3], (const float*)d_in[7],
                         (const float*)d_in[11], (const float*)d_in[15]};
  const float* bng[4] = {(const float*)d_in[4], (const float*)d_in[8],
                         (const float*)d_in[12], (const float*)d_in[16]};
  const float* bnb[4] = {(const float*)d_in[5], (const float*)d_in[9],
                         (const float*)d_in[13], (const float*)d_in[17]};
  const float* gw[4] = {(const float*)d_in[18], (const float*)d_in[20],
                        (const float*)d_in[22], (const float*)d_in[24]};
  const float* gb[4] = {(const float*)d_in[19], (const float*)d_in[21],
                        (const float*)d_in[23], (const float*)d_in[25]};
  const float* fw1 = (const float*)d_in[26]; const float* fb1 = (const float*)d_in[27];
  const float* fw2 = (const float*)d_in[28]; const float* fb2 = (const float*)d_in[29];
  const float* fw3 = (const float*)d_in[30]; const float* fb3 = (const float*)d_in[31];

  char* ws = (char*)d_ws;
  size_t off = 0;
  auto alloc = [&](size_t bytes) {
    size_t o = off; off += (bytes + 255) & ~(size_t)255; return o;
  };

  size_t o_x1 = alloc(512UL * 24 * 38 * 38 * 4);   // 71 MB; reused for P later
  size_t o_x2 = alloc(512UL * 24 * 19 * 19 * 4);
  size_t o_x3 = alloc(512UL * 24 * 10 * 10 * 4);
  size_t o_x4 = alloc(512UL * 24 * 5 * 5 * 4);
  size_t o_st = alloc(24 * 2 * 4);
  size_t o_wg1 = alloc(256UL * 64 * 2);
  size_t o_wg2 = alloc(256UL * 256 * 2);
  size_t o_wg3 = alloc(256UL * 256 * 2);
  size_t o_wg4 = alloc(256UL * 256 * 2);
  size_t o_wf1 = alloc(256UL * 256 * 2);
  size_t o_wf2 = alloc(256UL * 256 * 2);
  size_t o_H1 = alloc(320000UL * 256 * 2);         // 164 MB
  size_t o_H2 = alloc(320000UL * 256 * 2);         // 164 MB
  size_t o_xg = alloc(512UL * 256 * 2);
  size_t o_y1 = alloc(512UL * 256 * 2);
  size_t o_y2 = alloc(512UL * 256 * 2);
  size_t o_P  = o_x1;                              // overlay (41 MB <= 71 MB)

  // ---- conv stack ----
  struct LD { int Cin, Hin, Win, Hout, Wout; };
  LD ld[4] = {{3, 75, 75, 38, 38}, {24, 38, 38, 19, 19},
              {24, 19, 19, 10, 10}, {24, 10, 10, 5, 5}};
  float* xs[4] = {(float*)(ws + o_x1), (float*)(ws + o_x2),
                  (float*)(ws + o_x3), (float*)(ws + o_x4)};
  const float* cur = img;
  float* stats = (float*)(ws + o_st);
  for (int L = 0; L < 4; ++L) {
    int HW = ld[L].Hout * ld[L].Wout;
    int total = MBB * 24 * HW;
    conv_relu_k<<<(total + 255) / 256, 256, 0, stream>>>(
        cur, cw[L], cb[L], xs[L], ld[L].Cin, ld[L].Hin, ld[L].Win, ld[L].Hout, ld[L].Wout);
    bn_stats_k<<<24, 256, 0, stream>>>(xs[L], stats, HW);
    bn_apply_k<<<(total + 255) / 256, 256, 0, stream>>>(xs[L], stats, bng[L], bnb[L], HW);
    cur = xs[L];
  }

  // ---- weights -> bf16 (gw1 padded 63->64) ----
  unsigned short* wg1 = (unsigned short*)(ws + o_wg1);
  unsigned short* wg2 = (unsigned short*)(ws + o_wg2);
  unsigned short* wg3 = (unsigned short*)(ws + o_wg3);
  unsigned short* wg4 = (unsigned short*)(ws + o_wg4);
  unsigned short* wf1b = (unsigned short*)(ws + o_wf1);
  unsigned short* wf2b = (unsigned short*)(ws + o_wf2);
  wconv_k<<<(256 * 64 + 255) / 256, 256, 0, stream>>>(gw[0], wg1, 256, 63, 64);
  wconv_k<<<256, 256, 0, stream>>>(gw[1], wg2, 256, 256, 256);
  wconv_k<<<256, 256, 0, stream>>>(gw[2], wg3, 256, 256, 256);
  wconv_k<<<256, 256, 0, stream>>>(gw[3], wg4, 256, 256, 256);
  wconv_k<<<256, 256, 0, stream>>>(fw1, wf1b, 256, 256, 256);
  wconv_k<<<256, 256, 0, stream>>>(fw2, wf2b, 256, 256, 256);

  // ---- pair features ----
  unsigned short* P  = (unsigned short*)(ws + o_P);
  unsigned short* H1 = (unsigned short*)(ws + o_H1);
  unsigned short* H2 = (unsigned short*)(ws + o_H2);
  pair_k<<<(MBB * 625 * 64) / 256, 256, 0, stream>>>(xs[3], qst, P);

  // ---- g-network: 4 WMMA GEMMs over 320000 rows ----
  // grid: x = N blocks (fast), y = M blocks (slow) -> A tile reused from L2
  dim3 gg(256 / BN, 320000 / BM);   // (2, 2500)
  gemm_bias_relu<<<gg, 256, 0, stream>>>(P,  wg1, gb[0], H1, 320000, 256, 64, 1);
  gemm_bias_relu<<<gg, 256, 0, stream>>>(H1, wg2, gb[1], H2, 320000, 256, 256, 1);
  gemm_bias_relu<<<gg, 256, 0, stream>>>(H2, wg3, gb[2], H1, 320000, 256, 256, 1);
  gemm_bias_relu<<<gg, 256, 0, stream>>>(H1, wg4, gb[3], H2, 320000, 256, 256, 1);

  // ---- sum pool + f-network ----
  unsigned short* xg = (unsigned short*)(ws + o_xg);
  unsigned short* y1 = (unsigned short*)(ws + o_y1);
  unsigned short* y2 = (unsigned short*)(ws + o_y2);
  pool_k<<<(MBB * 256) / 256, 256, 0, stream>>>(H2, xg);

  dim3 gf(256 / BN, 512 / BM);      // (2, 4)
  gemm_bias_relu<<<gf, 256, 0, stream>>>(xg, wf1b, fb1, y1, 512, 256, 256, 1);
  gemm_bias_relu<<<gf, 256, 0, stream>>>(y1, wf2b, fb2, y2, 512, 256, 256, 1);

  final_k<<<MBB, 32, 0, stream>>>(y2, fw3, fb3, (float*)d_out);
}
